// MaskedAttention_38062000177660
// MI455X (gfx1250) — matrix operand order repaired and checked by reading, not compiled, not run on hardware
//
#include <hip/hip_runtime.h>
#include <stdint.h>

typedef __attribute__((ext_vector_type(16))) _Float16 v16h;
typedef __attribute__((ext_vector_type(8)))  float    v8f;
typedef __attribute__((ext_vector_type(2)))  _Float16 h2;
typedef __attribute__((ext_vector_type(4)))  uint32_t u32x4;
typedef __attribute__((ext_vector_type(8)))  uint32_t u32x8;

constexpr int B_ = 2, H_ = 12, S_ = 2048, D_ = 64;
constexpr int QW = 32;   // query rows per wave (two 16-row WMMA tiles)
constexpr int KT = 32;   // keys per inner iteration
constexpr int NW = 8;    // waves per block (256 threads, wave32)
constexpr int KPAD = 66; // 64 halves + 1 DW pad (TDM pad_interval=32DW, pad=1DW)
constexpr int VPAD = 34; // 32 halves + 1 DW pad (TDM pad_interval=16DW, pad=1DW)
constexpr int PPAD = 34;

// ---------------- one-time f32 -> f16 precision pass ------------------------
__global__ __launch_bounds__(256)
void cvt_f16_rowmajor(const float* __restrict__ src, _Float16* __restrict__ dst,
                      int n4) {
    int i = blockIdx.x * blockDim.x + threadIdx.x;
    if (i < n4) {
        float4 f = ((const float4*)src)[i];
        _Float16* d = dst + (size_t)i * 4;
        d[0] = (_Float16)f.x; d[1] = (_Float16)f.y;
        d[2] = (_Float16)f.z; d[3] = (_Float16)f.w;
    }
}

__global__ __launch_bounds__(256)
void cvt_f16_transpose(const float* __restrict__ v, _Float16* __restrict__ vt) {
    // v: [bh][s][d] f32 -> vt: [bh][d][s] f16
    size_t i = (size_t)blockIdx.x * blockDim.x + threadIdx.x;
    int d      = (int)(i & (D_ - 1));
    size_t sd  = i >> 6;                 // bh*S + s
    int s      = (int)(sd & (S_ - 1));
    size_t bh  = sd >> 11;               // /S
    vt[(bh * D_ + d) * S_ + s] = (_Float16)v[i];
}

// ---------------- TDM: 2D tile global->LDS with row padding -----------------
__device__ __forceinline__
void tdm_load_2d(uint32_t lds_off, const _Float16* gaddr,
                 uint32_t tensor_d0, uint32_t tensor_d1,
                 uint32_t tile_d0, uint32_t tile_d1,
                 uint32_t stride0, uint32_t pad_interval_code) {
    uint64_t ga = (uint64_t)(uintptr_t)gaddr;
    u32x4 g0;
    g0[0] = 1u;                                   // count=1, user descriptor
    g0[1] = lds_off;                              // LDS byte address
    g0[2] = (uint32_t)ga;                         // global_addr[31:0]
    g0[3] = ((uint32_t)(ga >> 32) & 0x01FFFFFFu)  // global_addr[56:32]
            | (2u << 30);                         // type=2 ("image")
    u32x8 g1;
    g1[0] = (1u << 16)                            // data_size = 2 bytes
          | (1u << 20)                            // pad_enable
          | (pad_interval_code << 22);            // pad_amount code 0 = 1 DWORD
    g1[1] = (tensor_d0 & 0xFFFFu) << 16;          // tensor_dim0[15:0]
    g1[2] = (tensor_d0 >> 16) | ((tensor_d1 & 0xFFFFu) << 16);
    g1[3] = (tensor_d1 >> 16) | (tile_d0 << 16);  // tile_dim0
    g1[4] = tile_d1 & 0xFFFFu;                    // tile_dim1 (tile_dim2=0: 2D)
    g1[5] = stride0;                              // tensor_dim0_stride[31:0]
    g1[6] = 0;                                    // stride0[47:32]=0, d1_stride lo=0
    g1[7] = 0;
    asm volatile("tensor_load_to_lds %0, %1" :: "s"(g0), "s"(g1) : "memory");
}

// ---------------- main flash-attention kernel --------------------------------
__global__ __launch_bounds__(256)
void MaskedAttention_fa_wmma(const float* __restrict__ q,
                             const _Float16* __restrict__ kh,   // [bh][s][d] f16
                             const _Float16* __restrict__ vt,   // [bh][d][s] f16
                             const float* __restrict__ mask,
                             float* __restrict__ out)
{
    __shared__ _Float16 Kbuf[2][KT][KPAD];     // [key][d], double buffered
    __shared__ _Float16 Vbuf[2][D_][VPAD];     // [d][key], double buffered
    __shared__ _Float16 Psh[NW][2][16][PPAD];  // per-wave P staging (2 q-tiles)

    const int tid  = threadIdx.x;
    const int wave = tid >> 5;
    const int lane = tid & 31;
    const int n16  = lane & 15;
    const int hi   = lane >> 4;          // 0: lanes 0-15, 1: lanes 16-31

    const int bh = blockIdx.y;           // 0..B*H-1
    const int b  = bh / H_;
    const int qrow0 = blockIdx.x * (NW * QW) + wave * QW;

    const float*    qb = q    + (size_t)bh * S_ * D_;
    const _Float16* kb = kh   + (size_t)bh * S_ * D_;
    const _Float16* vb = vt   + (size_t)bh * D_ * S_;
    const float*    mb = mask + (size_t)b  * S_ * S_;
    float*          ob = out  + (size_t)bh * S_ * D_;

    const uint32_t koff[2] = { (uint32_t)(uintptr_t)&Kbuf[0][0][0],
                               (uint32_t)(uintptr_t)&Kbuf[1][0][0] };
    const uint32_t voff[2] = { (uint32_t)(uintptr_t)&Vbuf[0][0][0],
                               (uint32_t)(uintptr_t)&Vbuf[1][0][0] };

    // ---- Q fragments for both 16-row tiles (f32 global, converted once).
    //      A layout: lane M = lane%16; K = 16*hi + 2*v + p
    v16h aq[2][2];
    #pragma unroll
    for (int t = 0; t < 2; ++t) {
        const float* qrow = qb + (size_t)(qrow0 + t * 16 + n16) * D_;
        #pragma unroll
        for (int c = 0; c < 2; ++c) {
            #pragma unroll
            for (int vv = 0; vv < 8; ++vv) {
                int d = 32 * c + 16 * hi + 2 * vv;
                float2 f = *(const float2*)(qrow + d);
                aq[t][c][2 * vv]     = (_Float16)f.x;
                aq[t][c][2 * vv + 1] = (_Float16)f.y;
            }
        }
    }

    v8f oA[4] = {}, oB[4] = {};
    float mrA[8], lrA[8], mrB[8], lrB[8];
    #pragma unroll
    for (int r = 0; r < 8; ++r) {
        mrA[r] = -1e30f; lrA[r] = 0.0f;
        mrB[r] = -1e30f; lrB[r] = 0.0f;
    }

    constexpr int NIT = S_ / KT;

    // prologue: TDM-stage tile 0 into buffer 0 (one wave drives the DMA engine)
    if (wave == 0) {
        tdm_load_2d(koff[0], kb, D_, S_, D_, KT, D_, 4);          // K: 32x64 rows
        tdm_load_2d(voff[0], vb, S_, D_, KT, D_, S_, 3);          // V^T: 64x32 rows
    }

    for (int j = 0; j < NIT; ++j) {
        const int kt  = j * KT;
        const int cur = j & 1;

        if (wave == 0) __builtin_amdgcn_s_wait_tensorcnt(0);
        __syncthreads();   // tile j visible in LDS to all waves

        // stream next tile into the other buffer while we compute
        if (wave == 0 && j + 1 < NIT) {
            const int nxt = (j + 1) & 1;
            tdm_load_2d(koff[nxt], kb + (size_t)(kt + KT) * D_, D_, S_, D_, KT, D_, 4);
            tdm_load_2d(voff[nxt], vb + (kt + KT),              S_, D_, KT, D_, S_, 3);
        }

        // ---- S tiles = Q @ K^T : both q-tiles share each B fragment.
        v8f accA[2], accB[2];
        #pragma unroll
        for (int nh = 0; nh < 2; ++nh) {
            v16h bk0, bk1;  // B layout (contiguous k, same as A): lane N = lane%16; K = 16*hi + 2*v + p
            #pragma unroll
            for (int vv = 0; vv < 8; ++vv) {
                int kdx = 16 * hi + 2 * vv;
                h2 p0 = *(const h2*)&Kbuf[cur][nh * 16 + n16][0  + kdx];
                h2 p1 = *(const h2*)&Kbuf[cur][nh * 16 + n16][32 + kdx];
                bk0[2 * vv] = p0.x; bk0[2 * vv + 1] = p0.y;
                bk1[2 * vv] = p1.x; bk1[2 * vv + 1] = p1.y;
            }
            v8f cA = {}, cB = {};
            cA = __builtin_amdgcn_wmma_f32_16x16x32_f16(false, aq[0][0], false, bk0,
                                                        (short)0, cA, false, false);
            cA = __builtin_amdgcn_wmma_f32_16x16x32_f16(false, aq[0][1], false, bk1,
                                                        (short)0, cA, false, false);
            cB = __builtin_amdgcn_wmma_f32_16x16x32_f16(false, aq[1][0], false, bk0,
                                                        (short)0, cB, false, false);
            cB = __builtin_amdgcn_wmma_f32_16x16x32_f16(false, aq[1][1], false, bk1,
                                                        (short)0, cB, false, false);
            accA[nh] = cA;
            accB[nh] = cB;
        }

        // ---- mask add + online softmax + P staging (per 16-row tile)
        auto tile_softmax = [&](v8f* acc, float* mr, float* lr, v8f* o,
                                int tIdx, int qbase) {
            #pragma unroll
            for (int nh = 0; nh < 2; ++nh) {
                #pragma unroll
                for (int r = 0; r < 8; ++r) {
                    int qr = qbase + r + 8 * hi;
                    acc[nh][r] += mb[(size_t)qr * S_ + kt + nh * 16 + n16];
                }
            }
            #pragma unroll
            for (int r = 0; r < 8; ++r) {
                float tmax = fmaxf(acc[0][r], acc[1][r]);
                #pragma unroll
                for (int mset = 1; mset <= 8; mset <<= 1)
                    tmax = fmaxf(tmax, __shfl_xor(tmax, mset, 32));
                float mnew  = fmaxf(mr[r], tmax);
                float scale = __expf(mr[r] - mnew);
                float p0 = __expf(acc[0][r] - mnew);
                float p1 = __expf(acc[1][r] - mnew);
                float rs = p0 + p1;
                #pragma unroll
                for (int mset = 1; mset <= 8; mset <<= 1)
                    rs += __shfl_xor(rs, mset, 32);
                lr[r] = lr[r] * scale + rs;
                mr[r] = mnew;
                o[0][r] *= scale; o[1][r] *= scale;
                o[2][r] *= scale; o[3][r] *= scale;
                Psh[wave][tIdx][r + 8 * hi][0  + n16] = (_Float16)p0;
                Psh[wave][tIdx][r + 8 * hi][16 + n16] = (_Float16)p1;
            }
        };
        tile_softmax(accA, mrA, lrA, oA, 0, qrow0);
        tile_softmax(accB, mrB, lrB, oB, 1, qrow0 + 16);

        // same-wave LDS RAW (C-layout -> A-layout transpose): explicit DS wait
        asm volatile("s_wait_dscnt 0" ::: "memory");

        v16h apA, apB;
        #pragma unroll
        for (int vv = 0; vv < 8; ++vv) {
            int kk = 16 * hi + 2 * vv;
            h2 pa = *(const h2*)&Psh[wave][0][n16][kk];
            h2 pb = *(const h2*)&Psh[wave][1][n16][kk];
            apA[2 * vv] = pa.x; apA[2 * vv + 1] = pa.y;
            apB[2 * vv] = pb.x; apB[2 * vv + 1] = pb.y;
        }

        // ---- O += P @ V : each V B-fragment feeds both q-tiles
        #pragma unroll
        for (int t = 0; t < 4; ++t) {
            v16h bv;
            #pragma unroll
            for (int vv = 0; vv < 8; ++vv) {
                h2 pv2 = *(const h2*)&Vbuf[cur][t * 16 + n16][16 * hi + 2 * vv];
                bv[2 * vv] = pv2.x; bv[2 * vv + 1] = pv2.y;
            }
            oA[t] = __builtin_amdgcn_wmma_f32_16x16x32_f16(false, apA, false, bv,
                                                           (short)0, oA[t], false, false);
            oB[t] = __builtin_amdgcn_wmma_f32_16x16x32_f16(false, apB, false, bv,
                                                           (short)0, oB[t], false, false);
        }

        __syncthreads();   // everyone done reading buf[cur] before TDM reuses it
    }

    // ---- normalize and write O (f32)
    #pragma unroll
    for (int r = 0; r < 8; ++r) {
        float invA = 1.0f / lrA[r];
        float invB = 1.0f / lrB[r];
        float* rowA = ob + (size_t)(qrow0 + r + 8 * hi) * D_;
        float* rowB = ob + (size_t)(qrow0 + 16 + r + 8 * hi) * D_;
        #pragma unroll
        for (int t = 0; t < 4; ++t) {
            rowA[t * 16 + n16] = oA[t][r] * invA;
            rowB[t * 16 + n16] = oB[t][r] * invB;
        }
    }
}

extern "C" void kernel_launch(void* const* d_in, const int* in_sizes, int n_in,
                              void* d_out, int out_size, void* d_ws, size_t ws_size,
                              hipStream_t stream) {
    (void)in_sizes; (void)n_in; (void)out_size; (void)ws_size;
    const float* q = (const float*)d_in[0];
    const float* k = (const float*)d_in[1];
    const float* v = (const float*)d_in[2];
    const float* m = (const float*)d_in[3];
    float* out = (float*)d_out;

    const size_t nElem = (size_t)B_ * H_ * S_ * D_;   // 3,145,728
    _Float16* kh = (_Float16*)d_ws;                   // f16 K, row-major
    _Float16* vt = kh + nElem;                        // f16 V, transposed [bh][d][s]

    cvt_f16_rowmajor<<<dim3((int)(nElem / 4 / 256)), dim3(256), 0, stream>>>(
        k, kh, (int)(nElem / 4));
    cvt_f16_transpose<<<dim3((int)(nElem / 256)), dim3(256), 0, stream>>>(v, vt);

    dim3 grid(S_ / (NW * QW), B_ * H_);   // (8, 24)
    MaskedAttention_fa_wmma<<<grid, dim3(256), 0, stream>>>(q, kh, vt, m, out);
}
